// SpatialMultiHeadAttention_79645873537455
// MI455X (gfx1250) — matrix operand order repaired and checked by reading, not compiled, not run on hardware
//
#include <hip/hip_runtime.h>

#define S_LEN  1024
#define DM     1024
#define HEADS  16
#define DK     64

typedef __bf16 bf16_t;
typedef __attribute__((ext_vector_type(16))) __bf16 v16bf;
typedef __attribute__((ext_vector_type(8)))  __bf16 v8bf;
typedef __attribute__((ext_vector_type(8)))  float  v8f;
typedef __attribute__((ext_vector_type(4)))  float  v4f;

__device__ __forceinline__ v8f wmma_bf16(v16bf a, v16bf b, v8f c) {
    // v_wmma_f32_16x16x32_bf16: D = A(16x32) * B(32x16) + C(16x16 f32)
    return __builtin_amdgcn_wmma_f32_16x16x32_bf16(
        /*neg_a=*/false, a, /*neg_b=*/false, b,
        /*c_mod=*/(short)0, c, /*reuse_a=*/false, /*reuse_b=*/false);
}

__device__ __forceinline__ v16bf cat8(v8bf lo, v8bf hi) {
    return __builtin_shufflevector(lo, hi, 0,1,2,3,4,5,6,7,8,9,10,11,12,13,14,15);
}

// ---------------------------------------------------------------------------
// Kernel 1: ContinuousPositionBias MLP evaluated once per distinct (dy,dx).
// table layout: [HEADS][63][63] float
// ---------------------------------------------------------------------------
__global__ __launch_bounds__(256)
void pos_bias_table_kernel(const float* __restrict__ w0, const float* __restrict__ b0,
                           const float* __restrict__ w1, const float* __restrict__ b1,
                           const float* __restrict__ w2, const float* __restrict__ b2,
                           float* __restrict__ table) {
    const int dx = (int)blockIdx.x - 31;
    const int dy = (int)blockIdx.y - 31;
    const int t  = threadIdx.x;                 // 0..255

    __shared__ float h1[256];
    __shared__ float h2[256];

    // layer 0: in=2 -> 256 ; w0 is [2][256] row-major (y = x @ W)
    float v = (float)dy * w0[t] + (float)dx * w0[256 + t] + b0[t];
    h1[t] = (v > 0.0f) ? v : 0.01f * v;
    __syncthreads();

    // layer 1: 256 -> 256
    float s = b1[t];
    #pragma unroll 8
    for (int i = 0; i < 256; ++i) s += h1[i] * w1[i * 256 + t];
    h2[t] = (s > 0.0f) ? s : 0.01f * s;
    __syncthreads();

    // layer 2: 256 -> 16
    if (t < HEADS) {
        float o = b2[t];
        #pragma unroll 8
        for (int i = 0; i < 256; ++i) o += h2[i] * w2[i * HEADS + t];
        table[t * 3969 + (dy + 31) * 63 + (dx + 31)] = o;
    }
}

// ---------------------------------------------------------------------------
// Kernel 2: QKV projections. X[4096x1024] fp32 @ W[1024x1024] fp32 -> bf16.
// grid (32, 8, 3): 128x128 output tile per WG, z selects Q/K/V.
// Q,K stored [B][H][S][DK]; V stored transposed [B][H][DK][S].
// ---------------------------------------------------------------------------
__global__ __launch_bounds__(256)
void proj_kernel(const float* __restrict__ q_in, const float* __restrict__ k_in,
                 const float* __restrict__ v_in,
                 const float* __restrict__ wq, const float* __restrict__ bq,
                 const float* __restrict__ wk, const float* __restrict__ bk,
                 const float* __restrict__ wv, const float* __restrict__ bv,
                 bf16_t* __restrict__ Qb, bf16_t* __restrict__ Kb,
                 bf16_t* __restrict__ Vt) {
    const int z = blockIdx.z;
    const float* X  = (z == 0) ? q_in : (z == 1) ? k_in : v_in;
    const float* W  = (z == 0) ? wq   : (z == 1) ? wk   : wv;
    const float* Bs = (z == 0) ? bq   : (z == 1) ? bk   : bv;

    __shared__ bf16_t Bt[128 * 40];   // B tile transposed [n][kd], stride 40 (pad)

    const int tid  = threadIdx.x;
    const int wave = tid >> 5, lane = tid & 31;
    const int half = lane >> 4, lm = lane & 15;
    const int row0 = blockIdx.x * 128, n0 = blockIdx.y * 128;
    const int arow = row0 + wave * 16 + lm;
    const float* ap = X + (size_t)arow * DM;

    v8f zero = {};
    v8f acc[8];
    #pragma unroll
    for (int t = 0; t < 8; ++t) acc[t] = zero;

    for (int k0 = 0; k0 < DM; k0 += 32) {
        // ---- stage W[k0..k0+31][n0..n0+127] into LDS transposed as bf16 ----
        #pragma unroll
        for (int ii = 0; ii < 4; ++ii) {
            int i  = tid + ii * 256;        // 0..1023
            int kd = i >> 5, n4 = i & 31;   // kd 0..31, n4 covers 4 cols
            v4f w4 = *(const v4f*)(W + (size_t)(k0 + kd) * DM + n0 + 4 * n4);
            #pragma unroll
            for (int j = 0; j < 4; ++j)
                Bt[(4 * n4 + j) * 40 + kd] = (bf16_t)w4[j];
        }
        __syncthreads();

        // ---- A 16x32 tile in ISA A-layout (per-lane two 16B groups) ----
        const int ka = k0 + 8 * half;
        v4f f0 = *(const v4f*)(ap + ka);
        v4f f1 = *(const v4f*)(ap + ka + 4);
        v4f f2 = *(const v4f*)(ap + ka + 16);
        v4f f3 = *(const v4f*)(ap + ka + 20);
        v16bf A;
        #pragma unroll
        for (int j = 0; j < 4; ++j) {
            A[j]      = (bf16_t)f0[j];
            A[4 + j]  = (bf16_t)f1[j];
            A[8 + j]  = (bf16_t)f2[j];
            A[12 + j] = (bf16_t)f3[j];
        }

        #pragma unroll
        for (int t = 0; t < 8; ++t) {
            const v8bf* bp = (const v8bf*)(Bt + (16 * t + lm) * 40 + 8 * half);
            v16bf Bv = cat8(bp[0], bp[2]);
            acc[t] = wmma_bf16(A, Bv, acc[t]);
        }
        __syncthreads();
    }

    // ---- epilogue: +bias, split heads, write bf16 ----
    #pragma unroll
    for (int t = 0; t < 8; ++t) {
        const int col = n0 + 16 * t + lm;
        const int hh = col >> 6, d = col & 63;
        const float bcol = Bs[col];
        #pragma unroll
        for (int r = 0; r < 8; ++r) {
            const int row = row0 + wave * 16 + r + 8 * half;
            const int bb = row >> 10, ss = row & 1023;
            const float val = acc[t][r] + bcol;
            if (z == 2)
                Vt[(((size_t)(bb * HEADS + hh)) * DK + d) * S_LEN + ss] = (bf16_t)val;
            else if (z == 1)
                Kb[(((size_t)(bb * HEADS + hh)) * S_LEN + ss) * DK + d] = (bf16_t)val;
            else
                Qb[(((size_t)(bb * HEADS + hh)) * S_LEN + ss) * DK + d] = (bf16_t)val;
        }
    }
}

// ---------------------------------------------------------------------------
// Kernel 3: flash attention with LDS-resident relative-position bias table.
// grid (8, 16, 4) = (S/128, H, B), 256 threads = 8 waves, 16 q-rows per wave.
// ---------------------------------------------------------------------------
__global__ __launch_bounds__(256)
void attn_kernel(const bf16_t* __restrict__ Qb, const bf16_t* __restrict__ Kb,
                 const bf16_t* __restrict__ Vt, const float* __restrict__ table,
                 bf16_t* __restrict__ Ctx) {
    __shared__ float  biasTab[63 * 63];
    __shared__ bf16_t pstage[8][16 * 40];   // per-wave P transpose buffer

    const int tid  = threadIdx.x;
    const int wave = tid >> 5, lane = tid & 31;
    const int half = lane >> 4, lm = lane & 15;
    const int h = blockIdx.y, b = blockIdx.z;
    const int q0 = blockIdx.x * 128 + wave * 16;

    for (int i = tid; i < 63 * 63; i += 256) biasTab[i] = table[h * 3969 + i];
    __syncthreads();

    const size_t bh = (size_t)(b * HEADS + h);
    const bf16_t* Qrow  = Qb + (bh * S_LEN + q0 + lm) * DK;
    const bf16_t* Kbase = Kb + bh * S_LEN * DK;
    const bf16_t* Vbase = Vt + bh * DK * S_LEN;
    bf16_t* prow = &pstage[wave][0];

    // Q tile in A-layout: A0 covers d 0..31, A1 covers d 32..63
    v16bf A0 = cat8(*(const v8bf*)(Qrow + 16 * half),
                    *(const v8bf*)(Qrow + 16 * half + 8));
    v16bf A1 = cat8(*(const v8bf*)(Qrow + 32 + 16 * half),
                    *(const v8bf*)(Qrow + 32 + 16 * half + 8));

    int qy[8], qx[8];
    #pragma unroll
    for (int r = 0; r < 8; ++r) {
        const int qq = q0 + r + 8 * half;
        qy[r] = qq >> 5; qx[r] = qq & 31;
    }

    float mrow[8], lrow[8];
    #pragma unroll
    for (int r = 0; r < 8; ++r) { mrow[r] = -1e30f; lrow[r] = 0.0f; }
    v8f zero = {};
    v8f acc[4];
    #pragma unroll
    for (int t = 0; t < 4; ++t) acc[t] = zero;

    for (int j = 0; j < S_LEN; j += 32) {
        if (j + 32 < S_LEN) {
            __builtin_prefetch(Kbase + (size_t)(j + 32 + lm) * DK, 0, 3);
            __builtin_prefetch(Vbase + (size_t)lm * S_LEN + j + 32, 0, 3);
        }
        // ---- scores: two 16x16 tiles (keys j..j+15, j+16..j+31) ----
        const bf16_t* kp0 = Kbase + (size_t)(j + lm) * DK + 16 * half;
        const bf16_t* kp1 = Kbase + (size_t)(j + 16 + lm) * DK + 16 * half;
        v8f s0 = zero, s1 = zero;
        s0 = wmma_bf16(A0, *(const v16bf*)kp0,        s0);
        s0 = wmma_bf16(A1, *(const v16bf*)(kp0 + 32), s0);
        s1 = wmma_bf16(A0, *(const v16bf*)kp1,        s1);
        s1 = wmma_bf16(A1, *(const v16bf*)(kp1 + 32), s1);

        const int k0i = j + lm,      k1i = j + 16 + lm;
        const int ky0 = k0i >> 5, kx0 = k0i & 31;
        const int ky1 = k1i >> 5, kx1 = k1i & 31;

        float p0[8], p1[8], sr[8];
        #pragma unroll
        for (int r = 0; r < 8; ++r) {
            float t0 = s0[r] * 0.125f + biasTab[(qy[r] - ky0 + 31) * 63 + (qx[r] - kx0 + 31)];
            float t1 = s1[r] * 0.125f + biasTab[(qy[r] - ky1 + 31) * 63 + (qx[r] - kx1 + 31)];
            // row max across the 16 lanes of this half-wave
            float rm = fmaxf(t0, t1);
            rm = fmaxf(rm, __shfl_xor(rm, 1, 32));
            rm = fmaxf(rm, __shfl_xor(rm, 2, 32));
            rm = fmaxf(rm, __shfl_xor(rm, 4, 32));
            rm = fmaxf(rm, __shfl_xor(rm, 8, 32));
            const float mnew  = fmaxf(mrow[r], rm);
            const float scale = __expf(mrow[r] - mnew);
            mrow[r] = mnew;
            p0[r] = __expf(t0 - mnew);
            p1[r] = __expf(t1 - mnew);
            float ps = p0[r] + p1[r];
            ps += __shfl_xor(ps, 1, 32);
            ps += __shfl_xor(ps, 2, 32);
            ps += __shfl_xor(ps, 4, 32);
            ps += __shfl_xor(ps, 8, 32);
            lrow[r] = lrow[r] * scale + ps;
            sr[r] = scale;
        }
        #pragma unroll
        for (int t = 0; t < 4; ++t)
            #pragma unroll
            for (int r = 0; r < 8; ++r)
                acc[t][r] *= sr[r];

        // ---- transpose P (C-layout -> A-layout) through per-wave LDS ----
        #pragma unroll
        for (int r = 0; r < 8; ++r) {
            const int pr = r + 8 * half;
            prow[pr * 40 + lm]      = (bf16_t)p0[r];
            prow[pr * 40 + 16 + lm] = (bf16_t)p1[r];
        }
        // same-wave LDS ops are in-order; compiler inserts s_wait_dscnt
        v16bf PA = cat8(*(const v8bf*)(prow + lm * 40 + 16 * half),
                        *(const v8bf*)(prow + lm * 40 + 16 * half + 8));

        // ---- ctx += P (16x32) @ V (32x64), V stored transposed ----
        #pragma unroll
        for (int t = 0; t < 4; ++t) {
            const bf16_t* vp = Vbase + (size_t)(16 * t + lm) * S_LEN + j + 16 * half;
            acc[t] = wmma_bf16(PA, *(const v16bf*)vp, acc[t]);
        }
    }

    // ---- normalize and write ctx as bf16 [B*S][DM] (heads re-merged) ----
    #pragma unroll
    for (int r = 0; r < 8; ++r) lrow[r] = 1.0f / lrow[r];
    #pragma unroll
    for (int t = 0; t < 4; ++t) {
        const int col = h * DK + 16 * t + lm;
        #pragma unroll
        for (int r = 0; r < 8; ++r) {
            const int row = b * S_LEN + q0 + r + 8 * half;
            Ctx[(size_t)row * DM + col] = (bf16_t)(acc[t][r] * lrow[r]);
        }
    }
}

// ---------------------------------------------------------------------------
// Kernel 4: output projection. ctx(bf16)[4096x1024] @ wo(fp32->bf16) + bo -> f32
// ---------------------------------------------------------------------------
__global__ __launch_bounds__(256)
void outproj_kernel(const bf16_t* __restrict__ Ctx, const float* __restrict__ Wo,
                    const float* __restrict__ bo, float* __restrict__ out) {
    __shared__ bf16_t Bt[128 * 40];

    const int tid  = threadIdx.x;
    const int wave = tid >> 5, lane = tid & 31;
    const int half = lane >> 4, lm = lane & 15;
    const int row0 = blockIdx.x * 128, n0 = blockIdx.y * 128;
    const int arow = row0 + wave * 16 + lm;
    const bf16_t* ap = Ctx + (size_t)arow * DM;

    v8f zero = {};
    v8f acc[8];
    #pragma unroll
    for (int t = 0; t < 8; ++t) acc[t] = zero;

    for (int k0 = 0; k0 < DM; k0 += 32) {
        #pragma unroll
        for (int ii = 0; ii < 4; ++ii) {
            int i  = tid + ii * 256;
            int kd = i >> 5, n4 = i & 31;
            v4f w4 = *(const v4f*)(Wo + (size_t)(k0 + kd) * DM + n0 + 4 * n4);
            #pragma unroll
            for (int j = 0; j < 4; ++j)
                Bt[(4 * n4 + j) * 40 + kd] = (bf16_t)w4[j];
        }
        __syncthreads();

        v16bf A = cat8(*(const v8bf*)(ap + k0 + 8 * half),
                       *(const v8bf*)(ap + k0 + 8 * half + 16));

        #pragma unroll
        for (int t = 0; t < 8; ++t) {
            const v8bf* bp = (const v8bf*)(Bt + (16 * t + lm) * 40 + 8 * half);
            v16bf Bv = cat8(bp[0], bp[2]);
            acc[t] = wmma_bf16(A, Bv, acc[t]);
        }
        __syncthreads();
    }

    #pragma unroll
    for (int t = 0; t < 8; ++t) {
        const int col = n0 + 16 * t + lm;
        const float bcol = bo[col];
        #pragma unroll
        for (int r = 0; r < 8; ++r) {
            const int row = row0 + wave * 16 + r + 8 * half;
            out[(size_t)row * DM + col] = acc[t][r] + bcol;
        }
    }
}

// ---------------------------------------------------------------------------
extern "C" void kernel_launch(void* const* d_in, const int* in_sizes, int n_in,
                              void* d_out, int out_size, void* d_ws, size_t ws_size,
                              hipStream_t stream) {
    const float* query = (const float*)d_in[0];
    const float* key_  = (const float*)d_in[1];
    const float* value = (const float*)d_in[2];
    const float* wq = (const float*)d_in[3];  const float* bq = (const float*)d_in[4];
    const float* wk = (const float*)d_in[5];  const float* bk = (const float*)d_in[6];
    const float* wv = (const float*)d_in[7];  const float* bv = (const float*)d_in[8];
    const float* wo = (const float*)d_in[9];  const float* bo = (const float*)d_in[10];
    const float* pw0 = (const float*)d_in[11]; const float* pb0 = (const float*)d_in[12];
    const float* pw1 = (const float*)d_in[13]; const float* pb1 = (const float*)d_in[14];
    const float* pw2 = (const float*)d_in[15]; const float* pb2 = (const float*)d_in[16];

    const size_t MiB = 1u << 20;
    char* ws = (char*)d_ws;
    bf16_t* Qb    = (bf16_t*)(ws);             //  8 MiB: [B][H][S][DK]
    bf16_t* Kb    = (bf16_t*)(ws + 8 * MiB);   //  8 MiB: [B][H][S][DK]
    bf16_t* Vt    = (bf16_t*)(ws + 16 * MiB);  //  8 MiB: [B][H][DK][S]
    bf16_t* Ctx   = (bf16_t*)(ws + 24 * MiB);  //  8 MiB: [B*S][DM]
    float*  table = (float*)(ws + 32 * MiB);   //  254 KB: [H][63][63]

    pos_bias_table_kernel<<<dim3(63, 63), 256, 0, stream>>>(
        pw0, pb0, pw1, pb1, pw2, pb2, table);

    proj_kernel<<<dim3(32, 8, 3), 256, 0, stream>>>(
        query, key_, value, wq, bq, wk, bk, wv, bv, Qb, Kb, Vt);

    attn_kernel<<<dim3(S_LEN / 128, HEADS, 4), 256, 0, stream>>>(
        Qb, Kb, Vt, table, Ctx);

    outproj_kernel<<<dim3(32, 8), 256, 0, stream>>>(
        Ctx, wo, bo, (float*)d_out);
}
